// Linear_12773232738580
// MI455X (gfx1250) — matrix operand order repaired and checked by reading, not compiled, not run on hardware
//
#include <hip/hip_runtime.h>

typedef __attribute__((ext_vector_type(16))) _Float16 v16h;
typedef __attribute__((ext_vector_type(8)))  float    v8f;
typedef __attribute__((ext_vector_type(4)))  _Float16 v4h;
typedef __attribute__((ext_vector_type(4)))  float    v4f;

namespace {
constexpr int kCIn  = 128;
constexpr int kCOut = 256;
constexpr int kKN   = 3;
constexpr int kH    = 256;
constexpr int kW    = 256;
constexpr int kOH   = kH - kKN + 1;   // 254
constexpr int kOW   = kW - kKN + 1;   // 254
constexpr int kNPix = kH * kW;        // 65536
constexpr int kKSteps = (kKN * kKN) * (kCIn / 32);  // 9 taps * 4 chunks = 36
}  // namespace

// ---------------------------------------------------------------------------
// Preprocess 1: x (f32) -> xh (f16), elementwise, 4 elems/thread.
// ---------------------------------------------------------------------------
__global__ void cvt_x_f16_kernel(const float* __restrict__ x,
                                 _Float16* __restrict__ xh) {
  int i = (blockIdx.x * blockDim.x + threadIdx.x) << 2;
  v4f v = *(const v4f*)(x + i);
  v4h h;
  h.x = (_Float16)v.x;
  h.y = (_Float16)v.y;
  h.z = (_Float16)v.z;
  h.w = (_Float16)v.w;
  *(v4h*)(xh + i) = h;
}

// ---------------------------------------------------------------------------
// Preprocess 2: int32 weights -> f16 packed in WMMA B-fragment-native layout.
//   wf[kstep][ntg][lane][j]  (f16), linear t = ((kstep*16+ntg)*32+lane)*16+j
//   B 32x16 layout (wave32): lane = (n&15) + 16*((k>>3)&1), j = (k&7) + 8*(k>>4).
//   K reorder: kstep = kidx*4 + c0/32 ; original row = c*9 + kidx.
// ---------------------------------------------------------------------------
__global__ void build_wfrag_kernel(const int* __restrict__ wq,
                                   _Float16* __restrict__ wf) {
  int t = blockIdx.x * blockDim.x + threadIdx.x;  // 0 .. 36*16*32*16-1
  int j     = t & 15;
  int lane  = (t >> 4) & 31;
  int ntg   = (t >> 9) & 15;
  int kstep = t >> 13;
  int kidx  = kstep >> 2;
  int c0    = (kstep & 3) << 5;
  int krel  = ((lane >> 4) << 3) + (j & 7) + ((j >> 3) << 4);  // 0..31
  int c     = c0 + krel;
  int n     = (ntg << 4) + (lane & 15);
  wf[t] = (_Float16)wq[(c * 9 + kidx) * kCOut + n];
}

// ---------------------------------------------------------------------------
// Main kernel: implicit-GEMM conv via V_WMMA_F32_16X16X32_F16.
//   grid = (254, 2, 2): oi, oj-tile(128), n-half(128)
//   block = 256 threads = 8 waves; wave (wm, wn) owns 32 M x 64 N.
//   A tiles: GLOBAL_LOAD_ASYNC_TO_LDS_B64 scatter into fragment layout,
//            double-buffered LDS; per-lane source address maintained
//            incrementally (+64 B per step, +64832 B at the two ki steps).
//   B tiles: software-pipelined one K-step ahead in registers, plus a single
//            WGP-scope global_prefetch_b8 per step (lane*128 covers 4 KB).
// ---------------------------------------------------------------------------
__global__ void __launch_bounds__(256)
conv_gemm_wmma_kernel(const _Float16* __restrict__ xh,
                      const _Float16* __restrict__ wf,
                      const float* __restrict__ kscale,
                      const float* __restrict__ bias,
                      float* __restrict__ out) {
  // A staged in fragment-native layout: [buf][mtile(8)][lane(32)][j(16)] f16
  __shared__ __align__(32) _Float16 sA[2][8 * 32 * 16];  // 2 x 8 KB

  const int oi  = blockIdx.x;        // output row
  const int ojb = blockIdx.y << 7;   // output col base (0 / 128)
  const int n0  = blockIdx.z << 7;   // output channel base (0 / 128)

  const int tid  = threadIdx.x;
  const int lane = tid & 31;
  const int wv   = tid >> 5;
  const int wm   = wv & 3;           // wave M offset = wm*32
  const int wn   = wv >> 2;          // wave N offset = wn*64

  v8f acc[2][4];
  {
    v8f zero = {0.f, 0.f, 0.f, 0.f, 0.f, 0.f, 0.f, 0.f};
#pragma unroll
    for (int i = 0; i < 2; ++i)
#pragma unroll
      for (int j = 0; j < 4; ++j) acc[i][j] = zero;
  }

  // ---- A staging state (incremental addressing) ----
  // Thread handles pixel m = (tid>>3) + 32*it, channel chunk k = 4*(tid&7).
  // Global source for K-step 0 (ki=kj=0, c0=0); chunks it are +8192 B apart.
  // Step s -> s+1 moves the base by a scalar: +64 B (chunk/kj steps, since
  // pixel +256 B and c0 reset -192 B cancel) or +64832 B (ki steps s=11,23).
  // No clamp needed: max overshoot is <512 B past xh, which is still inside
  // the workspace (wfrag follows), and only feeds discarded oj>=254 outputs.
  unsigned long long ga =
      (unsigned long long)(size_t)xh +
      ((unsigned long long)((oi * kW + ojb + (tid >> 3)) * kCIn +
                            ((tid & 7) << 2))
       << 1);
  // Per-lane LDS destination swizzle (A-fragment layout), loop-invariant:
  // lane l holds row M=l&15, K=(l>>4)*8+{0..7,16..23}; slot j: k=(l>>4)*8+
  // (j<8?j:j+8).  tile=(m>>4) -> +1024 B; buf -> +8192 B; it -> +2048 B.
  const int l2 = ((tid >> 3) & 15) + (((tid >> 1) & 1) << 4);
  const int j0 = ((tid & 1) << 2) + (((tid >> 2) & 1) << 3);
  const unsigned lds0 = (unsigned)(size_t)(&sA[0][0]) +
                        (((tid >> 7) & 1) << 10) + (l2 << 5) + (j0 << 1);

  auto stage = [&](int buf) {
#pragma unroll
    for (int it = 0; it < 4; ++it) {
      unsigned lo = lds0 + (buf << 13) + (it << 11);
      unsigned long long g = ga + ((unsigned long long)it << 13);
      asm volatile("global_load_async_to_lds_b64 %0, %1, off"
                   :
                   : "v"(lo), "v"(g)
                   : "memory");
    }
  };
  auto advance = [&](int s) {  // move ga from staged step s to step s+1
    ga += (s == 11 || s == 23) ? 64832ull : 64ull;
  };

  const v16h* __restrict__ wfv = (const v16h*)wf;
  const int ntg0  = (n0 >> 4) + (wn << 2);       // first of 4 n-subtiles
  const int aoff0 = (wm << 10) + (lane << 4);    // f16 offset of A tile (wm*2)

  // Per-lane B fragment pointer for K-step 0; advances 512 v16h (16 KB)/step.
  const v16h* bp = wfv + (ntg0 << 5) + lane;
  // Per-lane prefetch base: wave region is 4 KB contiguous; lane*128 covers it.
  const unsigned long long pfb =
      (unsigned long long)(size_t)wf + ((unsigned long long)ntg0 << 10) +
      ((unsigned long long)lane << 7);

  auto loadB = [&](v16h (&bv)[4], const v16h* p) {
#pragma unroll
    for (int t = 0; t < 4; ++t) bv[t] = p[t << 5];
  };
  auto prefetchB = [&](int s) {
    if (s < kKSteps) {
      unsigned long long a = pfb + ((unsigned long long)s << 14);
      asm volatile("global_prefetch_b8 %0, off" : : "v"(a) : "memory");
    }
  };
  auto compute = [&](int buf, const v16h (&bv)[4]) {
    v16h a0 = *(const v16h*)(&sA[buf][aoff0]);
    v16h a1 = *(const v16h*)(&sA[buf][aoff0 + 512]);
#pragma unroll
    for (int t = 0; t < 4; ++t) {
      acc[0][t] = __builtin_amdgcn_wmma_f32_16x16x32_f16(
          false, a0, false, bv[t], (short)0, acc[0][t], false, false);
      acc[1][t] = __builtin_amdgcn_wmma_f32_16x16x32_f16(
          false, a1, false, bv[t], (short)0, acc[1][t], false, false);
    }
  };

  v16h b0[4], b1[4];
  loadB(b0, bp);            // B for step 0
  bp += 512;                // -> step 1
  stage(0);                 // A for step 0
  advance(0);
  asm volatile("s_wait_asynccnt 0x0" ::: "memory");
  __syncthreads();

  for (int ks = 0; ks < kKSteps; ks += 2) {
    // ---- phase A: compute step ks; fetch step ks+1 ----
    stage(1);               // ks+1 <= 35 always valid
    advance(ks + 1);
    loadB(b1, bp);
    bp += 512;
    prefetchB(ks + 2);
    compute(0, b0);
    asm volatile("s_wait_asynccnt 0x0" ::: "memory");
    __syncthreads();

    // ---- phase B: compute step ks+1; fetch step ks+2 ----
    if (ks + 2 < kKSteps) {
      stage(0);
      advance(ks + 2);
      loadB(b0, bp);
      bp += 512;
      prefetchB(ks + 3);
    }
    compute(1, b1);
    asm volatile("s_wait_asynccnt 0x0" ::: "memory");
    __syncthreads();
  }

  // Epilogue: D layout 16x16 f32: lane<16 -> N=lane, M=v; lane>=16 -> M=v+8.
  const float scale = kscale[0];
  const int hi = (lane >> 4) << 3;
  const int nl = lane & 15;
#pragma unroll
  for (int tm = 0; tm < 2; ++tm) {
#pragma unroll
    for (int tn = 0; tn < 4; ++tn) {
      const int n = n0 + (wn << 6) + (tn << 4) + nl;
      const float bv = bias[n];
#pragma unroll
      for (int v = 0; v < 8; ++v) {
        const int m  = (wm << 5) + (tm << 4) + hi + v;
        const int oj = ojb + m;
        if (oj < kOW) {
          out[(oi * kOW + oj) * kCOut + n] = acc[tm][tn][v] * scale + bv;
        }
      }
    }
  }
}

// ---------------------------------------------------------------------------
// Launcher
// ---------------------------------------------------------------------------
extern "C" void kernel_launch(void* const* d_in, const int* in_sizes, int n_in,
                              void* d_out, int out_size, void* d_ws, size_t ws_size,
                              hipStream_t stream) {
  const float* x      = (const float*)d_in[0];
  const int*   wq     = (const int*)d_in[1];
  const float* kscale = (const float*)d_in[2];
  const float* bias   = (const float*)d_in[3];
  float*       out    = (float*)d_out;

  // Workspace: xh (f16 copy of x) then wfrag (fragment-packed f16 weights).
  _Float16* xh = (_Float16*)d_ws;
  _Float16* wfrag = xh + (size_t)kNPix * kCIn;  // +16 MiB

  cvt_x_f16_kernel<<<(kNPix * kCIn) / (256 * 4), 256, 0, stream>>>(x, xh);
  build_wfrag_kernel<<<(kKSteps * 16 * 32 * 16) / 256, 256, 0, stream>>>(wq, wfrag);

  dim3 grid(kOH, (kOW + 127) / 128, kCOut / 128);
  conv_gemm_wmma_kernel<<<grid, 256, 0, stream>>>(xh, wfrag, kscale, bias, out);
}
